// RelaFusionLayer_3006477107421
// MI455X (gfx1250) — matrix operand order repaired and checked by reading, not compiled, not run on hardware
//
#include <hip/hip_runtime.h>
#include <hip/hip_bf16.h>

#define NN   512
#define DM   128
#define DFF_ 2048
#define HH   8

typedef __attribute__((ext_vector_type(16))) _Float16 v16h;
typedef __attribute__((ext_vector_type(8)))  _Float16 v8h;
typedef __attribute__((ext_vector_type(4)))  _Float16 v4h;
typedef __attribute__((ext_vector_type(8)))  float    v8f;

// ---------------- workspace layout (bytes) ----------------
static constexpr size_t SZ_FRAG  = (size_t)4*8*32*16*2;      // 32 KB per swizzled 128x128 weight
static constexpr size_t OFF_MEM16 = 0;                        // N*N*D f16 = 64 MB
static constexpr size_t OFF_WE   = OFF_MEM16 + (size_t)NN*NN*DM*2;
static constexpr size_t OFF_WP   = OFF_WE + SZ_FRAG;
static constexpr size_t OFF_NS   = OFF_WP + SZ_FRAG;          // node@w_mem_s  (N*D f32)
static constexpr size_t OFF_NT   = OFF_NS + (size_t)NN*DM*4;  // node@w_mem_t
static constexpr size_t OFF_QK   = OFF_NT + (size_t)NN*DM*4;  // folded q*wk  (N*H*D f32)
static constexpr size_t OFF_SB   = OFF_QK + (size_t)NN*HH*DM*4; // q·bk       (N*H f32)
static constexpr size_t OFF_SC   = OFF_SB + (size_t)NN*HH*4;  // scores/attn (N*H*N f32)
static constexpr size_t OFF_AM   = OFF_SC + (size_t)NN*HH*NN*4; // attn-weighted memory (N*H*D f32)

// ---------------- helpers ----------------
__device__ __forceinline__ float half16_sum(float v) {
  v += __shfl_xor(v, 1, 32);
  v += __shfl_xor(v, 2, 32);
  v += __shfl_xor(v, 4, 32);
  v += __shfl_xor(v, 8, 32);
  return v;   // reduced within each 16-lane half (rows stay in their half)
}

__device__ __forceinline__ float wave_sum32(float v) {
  v += __shfl_xor(v, 1, 32);
  v += __shfl_xor(v, 2, 32);
  v += __shfl_xor(v, 4, 32);
  v += __shfl_xor(v, 8, 32);
  v += __shfl_xor(v, 16, 32);
  return v;
}

__device__ __forceinline__ float block_sum_256(float v, float* red, int tid) {
  v = wave_sum32(v);
  __syncthreads();
  if ((tid & 31) == 0) red[tid >> 5] = v;
  __syncthreads();
  float s = 0.f;
#pragma unroll
  for (int i = 0; i < 8; ++i) s += red[i];
  return s;
}

// A: 16x128 f16 tile in LDS (row-major, 128-half row stride)
// B: pre-swizzled fragment buffer in global memory
// 4 K-steps x 8 column tiles = 32 v_wmma_f32_16x16x32_f16
__device__ __forceinline__ void gemm_tile(const _Float16* tile,
                                          const _Float16* __restrict__ wfrag,
                                          int lane, v8f acc[8]) {
#pragma unroll
  for (int t = 0; t < 4; ++t) {
    const int row = lane & 15;
    const int kb  = 32*t + ((lane & 16) ? 8 : 0);     // A 16-bit layout: K runs [b..b+7],[b+16..b+23]
    v8h lo = *(const v8h*)(tile + row*DM + kb);
    v8h hi = *(const v8h*)(tile + row*DM + kb + 16);
    v16h a;
#pragma unroll
    for (int u = 0; u < 8; ++u) { a[u] = lo[u]; a[8+u] = hi[u]; }
#pragma unroll
    for (int c = 0; c < 8; ++c) {
      v16h b = *(const v16h*)(wfrag + ((size_t)((t*8 + c)*32 + lane))*16);
      acc[c] = __builtin_amdgcn_wmma_f32_16x16x32_f16(
          false, a, false, b, (short)0, acc[c], false, false);
    }
  }
}

// ---------------- kernel 1: swizzle a 128x128 fp32 weight into f16 B-fragments ----------------
__global__ __launch_bounds__(256) void swz_kernel(const float* __restrict__ W,
                                                  _Float16* __restrict__ out) {
  int idx  = blockIdx.x*256 + threadIdx.x;   // 8192 threads, each writes one K-pair
  int v    = idx & 7;
  int lane = (idx >> 3) & 31;
  int c    = (idx >> 8) & 7;
  int t    = (idx >> 11);                    // 0..3
  int ncol = c*16 + (lane & 15);
  int k    = 32*t + ((lane & 16) ? 8 : 0) + 2*v + ((v & 4) ? 8 : 0);   // B 16-bit layout: K runs [b..b+7],[b+16..b+23], b = 8*(lane>=16)
  size_t o = ((size_t)((t*8 + c)*32 + lane))*16 + 2*v;
  out[o]   = (_Float16)W[(size_t)k*DM + ncol];
  out[o+1] = (_Float16)W[(size_t)(k+1)*DM + ncol];
}

// ---------------- kernel 2: node_s, node_t, folded qk, score bias ----------------
__global__ __launch_bounds__(128) void prep_kernel(
    const float* __restrict__ node,
    const float* __restrict__ w_mem_s, const float* __restrict__ w_mem_t,
    const float* __restrict__ wq, const float* __restrict__ bq,
    const float* __restrict__ wk, const float* __restrict__ bk,
    float* __restrict__ node_s, float* __restrict__ node_t,
    float* __restrict__ qk, float* __restrict__ sb) {
  const int n = blockIdx.x, tid = threadIdx.x;
  __shared__ float nrow[DM];
  __shared__ float qrow[DM];
  nrow[tid] = node[(size_t)n*DM + tid];
  __syncthreads();
  float as = 0.f, at = 0.f, aq = bq[tid];
  for (int k = 0; k < DM; ++k) {
    float x = nrow[k];
    as += x * w_mem_s[(size_t)k*DM + tid];
    at += x * w_mem_t[(size_t)k*DM + tid];
    aq += x * wq[(size_t)k*DM + tid];
  }
  node_s[(size_t)n*DM + tid] = as;
  node_t[(size_t)n*DM + tid] = at;
  qrow[tid] = aq;
  __syncthreads();
#pragma unroll
  for (int h = 0; h < HH; ++h) {
    float a = 0.f;
#pragma unroll
    for (int dh = 0; dh < 16; ++dh) a += wk[(size_t)tid*DM + h*16 + dh] * qrow[h*16 + dh];
    qk[((size_t)n*HH + h)*DM + tid] = a;
  }
  if (tid < HH) {
    float a = 0.f;
#pragma unroll
    for (int dh = 0; dh < 16; ++dh) a += bk[tid*16 + dh] * qrow[tid*16 + dh];
    sb[(size_t)n*HH + tid] = a;
  }
}

// ---------------- kernel 3: fused edge pipeline (2 WMMA GEMMs + 3 LayerNorms) ----------------
__global__ __launch_bounds__(128) void fuse_edge_kernel(
    const float* __restrict__ edge,
    const _Float16* __restrict__ wE, const _Float16* __restrict__ wP,
    const float* __restrict__ node_s, const float* __restrict__ node_t,
    const float* __restrict__ b_mem, const float* __restrict__ g_ln_mem, const float* __restrict__ b_ln_mem,
    const float* __restrict__ b_pe, const float* __restrict__ g_ln_pe, const float* __restrict__ b_ln_pe,
    const float* __restrict__ g_ln_edge, const float* __restrict__ b_ln_edge,
    _Float16* __restrict__ mem16, float* __restrict__ edge_out) {
  __shared__ __align__(16) float    sE[4][16*DM];   // fp32 edge tiles (residual path)
  __shared__ __align__(16) _Float16 sT[4][16*DM];   // f16 staging for WMMA A operands
  const int tid = threadIdx.x, w = tid >> 5, lane = tid & 31;
  const int r0 = blockIdx.x*64 + w*16;              // 16 consecutive flattened (i,j) rows
  float*    eT = sE[w];
  _Float16* tT = sT[w];
  const int ln  = lane & 15;
  const int hi8 = (lane & 16) ? 8 : 0;

  // load edge tile (16 x 128 fp32) -> LDS fp32 + f16
  const float* ep = edge + (size_t)r0*DM;
#pragma unroll
  for (int it = 0; it < 16; ++it) {
    int li = lane + it*32;                          // 512 float4 chunks
    float4 vv = *(const float4*)(ep + (size_t)li*4);
    *(float4*)(eT + li*4) = vv;
    tT[li*4+0] = (_Float16)vv.x;
    tT[li*4+1] = (_Float16)vv.y;
    tT[li*4+2] = (_Float16)vv.z;
    tT[li*4+3] = (_Float16)vv.w;
  }
  __syncthreads();

  const v8f zero = {0.f,0.f,0.f,0.f,0.f,0.f,0.f,0.f};
  v8f acc[8];
#pragma unroll
  for (int c = 0; c < 8; ++c) acc[c] = zero;

  // GEMM1: mem_pre = edge @ w_mem_e
  gemm_tile(tT, wE, lane, acc);

  // + node_s[j] + node_t[i] + b_mem
  const int ii = r0 >> 9;
  const int jb = r0 & (NN - 1);
#pragma unroll
  for (int c = 0; c < 8; ++c) {
    int f = c*16 + ln;
    float cb = b_mem[f] + node_t[(size_t)ii*DM + f];
#pragma unroll
    for (int g = 0; g < 8; ++g) {
      int j = jb + hi8 + g;
      acc[c][g] += node_s[(size_t)j*DM + f] + cb;
    }
  }
  // LN + ReLU -> memory, keep fp32 in acc, write f16 tile to LDS
  {
    float gl[8], bl[8];
#pragma unroll
    for (int c = 0; c < 8; ++c) { gl[c] = g_ln_mem[c*16+ln]; bl[c] = b_ln_mem[c*16+ln]; }
#pragma unroll
    for (int g = 0; g < 8; ++g) {
      float s = 0.f, q = 0.f;
#pragma unroll
      for (int c = 0; c < 8; ++c) { float v = acc[c][g]; s += v; q += v*v; }
      s = half16_sum(s); q = half16_sum(q);
      float m  = s * (1.f/128.f);
      float rs = rsqrtf(q*(1.f/128.f) - m*m + 1e-5f);
#pragma unroll
      for (int c = 0; c < 8; ++c) {
        float v = (acc[c][g] - m)*rs*gl[c] + bl[c];
        v = v > 0.f ? v : 0.f;
        acc[c][g] = v;
        tT[(hi8+g)*DM + c*16 + ln] = (_Float16)v;
      }
    }
  }
  __syncthreads();

  // store memory tile (f16) to global for the attention passes
  {
    _Float16* mp = mem16 + (size_t)r0*DM;
#pragma unroll
    for (int it = 0; it < 8; ++it) {
      int li = lane + it*32;                        // 256 v8h chunks
      *(v8h*)(mp + li*8) = *(const v8h*)(tT + li*8);
    }
  }

  // GEMM2: memory @ w_pe
#pragma unroll
  for (int c = 0; c < 8; ++c) acc[c] = zero;
  gemm_tile(tT, wP, lane, acc);
#pragma unroll
  for (int c = 0; c < 8; ++c) {
    float bp = b_pe[c*16 + ln];
#pragma unroll
    for (int g = 0; g < 8; ++g) acc[c][g] += bp;
  }
  // LN(pe) + ReLU, add fp32 edge residual
  {
    float gl[8], bl[8];
#pragma unroll
    for (int c = 0; c < 8; ++c) { gl[c] = g_ln_pe[c*16+ln]; bl[c] = b_ln_pe[c*16+ln]; }
#pragma unroll
    for (int g = 0; g < 8; ++g) {
      float s = 0.f, q = 0.f;
#pragma unroll
      for (int c = 0; c < 8; ++c) { float v = acc[c][g]; s += v; q += v*v; }
      s = half16_sum(s); q = half16_sum(q);
      float m  = s * (1.f/128.f);
      float rs = rsqrtf(q*(1.f/128.f) - m*m + 1e-5f);
#pragma unroll
      for (int c = 0; c < 8; ++c) {
        float u = (acc[c][g] - m)*rs*gl[c] + bl[c];
        u = u > 0.f ? u : 0.f;
        acc[c][g] = eT[(hi8+g)*DM + c*16 + ln] + u;
      }
    }
  }
  // LN(edge) -> edge_new
  {
    float gl[8], bl[8];
#pragma unroll
    for (int c = 0; c < 8; ++c) { gl[c] = g_ln_edge[c*16+ln]; bl[c] = b_ln_edge[c*16+ln]; }
#pragma unroll
    for (int g = 0; g < 8; ++g) {
      float s = 0.f, q = 0.f;
#pragma unroll
      for (int c = 0; c < 8; ++c) { float v = acc[c][g]; s += v; q += v*v; }
      s = half16_sum(s); q = half16_sum(q);
      float m  = s * (1.f/128.f);
      float rs = rsqrtf(q*(1.f/128.f) - m*m + 1e-5f);
#pragma unroll
      for (int c = 0; c < 8; ++c)
        eT[(hi8+g)*DM + c*16 + ln] = (acc[c][g] - m)*rs*gl[c] + bl[c];
    }
  }
  __syncthreads();
  {
    float* op = edge_out + (size_t)r0*DM;
#pragma unroll
    for (int it = 0; it < 16; ++it) {
      int li = lane + it*32;
      *(float4*)(op + (size_t)li*4) = *(const float4*)(eT + li*4);
    }
  }
}

// ---------------- kernel 4: attention scores (wk folded into q) ----------------
__global__ __launch_bounds__(256) void scores_kernel(
    const _Float16* __restrict__ mem16, const float* __restrict__ qk,
    const float* __restrict__ sb, const unsigned char* __restrict__ emask,
    float* __restrict__ sc) {
  const int n = blockIdx.x;
  const int tid = threadIdx.x, wid = tid >> 5, lane = tid & 31;
  __shared__ float qkl[HH*DM];
  __shared__ float sbl[HH];
  for (int idx = tid; idx < HH*DM; idx += 256) qkl[idx] = qk[(size_t)n*HH*DM + idx];
  if (tid < HH) sbl[tid] = sb[(size_t)n*HH + tid];
  __syncthreads();
  const int d0 = lane*4;
  for (int s = wid; s < NN; s += 8) {
    v4h m4 = *(const v4h*)(mem16 + ((size_t)s*NN + n)*DM + d0);
    float m0 = (float)m4[0], m1 = (float)m4[1], m2 = (float)m4[2], m3 = (float)m4[3];
    float r[HH];
#pragma unroll
    for (int h = 0; h < HH; ++h) {
      const float* qp = qkl + h*DM + d0;
      r[h] = m0*qp[0] + m1*qp[1] + m2*qp[2] + m3*qp[3];
    }
#pragma unroll
    for (int h = 0; h < HH; ++h) r[h] = wave_sum32(r[h]);
    if (lane == 0) {
      bool mk = emask[(size_t)n*NN + s] != 0;
#pragma unroll
      for (int h = 0; h < HH; ++h) {
        float v = (r[h] + sbl[h]) * 0.25f;          // 1/sqrt(DH), DH=16
        if (mk) v = -3.402823466e38f;               // finfo(f32).min
        sc[((size_t)n*HH + h)*NN + s] = v;
      }
    }
  }
}

// ---------------- kernel 5: softmax over s (in place) ----------------
__global__ __launch_bounds__(256) void softmax_kernel(float* __restrict__ sc) {
  const size_t row = blockIdx.x;
  const int tid = threadIdx.x;
  float* p = sc + row*NN;
  float a = p[tid], b = p[tid + 256];
  __shared__ float red[8];
  float mx = fmaxf(a, b);
  mx = fmaxf(mx, __shfl_xor(mx, 1, 32));
  mx = fmaxf(mx, __shfl_xor(mx, 2, 32));
  mx = fmaxf(mx, __shfl_xor(mx, 4, 32));
  mx = fmaxf(mx, __shfl_xor(mx, 8, 32));
  mx = fmaxf(mx, __shfl_xor(mx, 16, 32));
  if ((tid & 31) == 0) red[tid >> 5] = mx;
  __syncthreads();
  float m = red[0];
#pragma unroll
  for (int i = 1; i < 8; ++i) m = fmaxf(m, red[i]);
  __syncthreads();
  float e0 = expf(a - m), e1 = expf(b - m);
  float s = wave_sum32(e0 + e1);
  if ((tid & 31) == 0) red[tid >> 5] = s;
  __syncthreads();
  float tot = 0.f;
#pragma unroll
  for (int i = 0; i < 8; ++i) tot += red[i];
  float inv = 1.f / tot;
  p[tid] = e0 * inv;
  p[tid + 256] = e1 * inv;
}

// ---------------- kernel 6: am[n,h,:] = sum_s attn[n,h,s] * memory[s,n,:] ----------------
__global__ __launch_bounds__(128) void am_kernel(const _Float16* __restrict__ mem16,
                                                 const float* __restrict__ attn,
                                                 float* __restrict__ am) {
  const int n = blockIdx.x, tid = threadIdx.x;
  __shared__ float al[HH*NN];
  for (int idx = tid; idx < HH*NN; idx += 128) al[idx] = attn[(size_t)n*HH*NN + idx];
  __syncthreads();
  float acc[HH];
#pragma unroll
  for (int h = 0; h < HH; ++h) acc[h] = 0.f;
  for (int s = 0; s < NN; ++s) {
    float mv = (float)mem16[((size_t)s*NN + n)*DM + tid];
#pragma unroll
    for (int h = 0; h < HH; ++h) acc[h] += al[h*NN + s] * mv;
  }
#pragma unroll
  for (int h = 0; h < HH; ++h) am[((size_t)n*HH + h)*DM + tid] = acc[h];
}

// ---------------- kernel 7: ctx (wv folded), out proj, LN2, FFN, LN3 ----------------
__global__ __launch_bounds__(256) void final_kernel(
    const float* __restrict__ am, const float* __restrict__ wv_, const float* __restrict__ bv,
    const float* __restrict__ wo, const float* __restrict__ bo,
    const float* __restrict__ node, const float* __restrict__ g2, const float* __restrict__ b2,
    const float* __restrict__ wf1, const float* __restrict__ bf1,
    const float* __restrict__ wf2, const float* __restrict__ bf2,
    const float* __restrict__ g3, const float* __restrict__ b3,
    float* __restrict__ xout) {
  const int n = blockIdx.x, tid = threadIdx.x;
  __shared__ float amL[HH*DM];
  __shared__ float ctxL[DM];
  __shared__ float x1L[DM];
  __shared__ float h1L[DFF_];
  __shared__ float red[8];

  for (int idx = tid; idx < HH*DM; idx += 256) amL[idx] = am[(size_t)n*HH*DM + idx];
  __syncthreads();
  if (tid < DM) {                                  // ctx[n, h*16+dh] = am[n,h,:] @ wv[:, tid] + bv
    float a = bv[tid];
    const int h = tid >> 4;
    for (int d = 0; d < DM; ++d) a += amL[h*DM + d] * wv_[(size_t)d*DM + tid];
    ctxL[tid] = a;
  }
  __syncthreads();
  float y = 0.f;
  if (tid < DM) {                                  // x' = ctx @ wo + bo ; residual
    float a = bo[tid];
    for (int c = 0; c < DM; ++c) a += ctxL[c] * wo[(size_t)c*DM + tid];
    y = node[(size_t)n*DM + tid] + a;
  }
  float s1 = block_sum_256(tid < DM ? y     : 0.f, red, tid);
  float s2 = block_sum_256(tid < DM ? y*y   : 0.f, red, tid);
  float mean = s1 * (1.f/128.f);
  float rs   = rsqrtf(s2*(1.f/128.f) - mean*mean + 1e-5f);
  if (tid < DM) x1L[tid] = (y - mean)*rs*g2[tid] + b2[tid];
  __syncthreads();
  // FFN up + ReLU
  float fa[8];
#pragma unroll
  for (int jj = 0; jj < 8; ++jj) fa[jj] = bf1[tid + jj*256];
  for (int k = 0; k < DM; ++k) {
    float xv = x1L[k];
#pragma unroll
    for (int jj = 0; jj < 8; ++jj) fa[jj] += xv * wf1[(size_t)k*DFF_ + tid + jj*256];
  }
#pragma unroll
  for (int jj = 0; jj < 8; ++jj) h1L[tid + jj*256] = fa[jj] > 0.f ? fa[jj] : 0.f;
  __syncthreads();
  float y2 = 0.f;
  if (tid < DM) {                                  // FFN down + residual
    float a = bf2[tid];
    for (int j = 0; j < DFF_; ++j) a += h1L[j] * wf2[(size_t)j*DM + tid];
    y2 = x1L[tid] + a;
  }
  float t1 = block_sum_256(tid < DM ? y2      : 0.f, red, tid);
  float t2 = block_sum_256(tid < DM ? y2*y2   : 0.f, red, tid);
  float m2  = t1 * (1.f/128.f);
  float rs2 = rsqrtf(t2*(1.f/128.f) - m2*m2 + 1e-5f);
  if (tid < DM) xout[(size_t)n*DM + tid] = (y2 - m2)*rs2*g3[tid] + b3[tid];
}

// ---------------- launch ----------------
extern "C" void kernel_launch(void* const* d_in, const int* in_sizes, int n_in,
                              void* d_out, int out_size, void* d_ws, size_t ws_size,
                              hipStream_t stream) {
  (void)in_sizes; (void)n_in; (void)out_size; (void)ws_size;
  const float* node      = (const float*)d_in[0];
  const float* edge      = (const float*)d_in[1];
  const unsigned char* emask = (const unsigned char*)d_in[2];
  const float* w_mem_e   = (const float*)d_in[3];
  const float* w_mem_s   = (const float*)d_in[4];
  const float* w_mem_t   = (const float*)d_in[5];
  const float* b_mem     = (const float*)d_in[6];
  const float* g_ln_mem  = (const float*)d_in[7];
  const float* b_ln_mem  = (const float*)d_in[8];
  const float* w_pe      = (const float*)d_in[9];
  const float* b_pe      = (const float*)d_in[10];
  const float* g_ln_pe   = (const float*)d_in[11];
  const float* b_ln_pe   = (const float*)d_in[12];
  const float* g_ln_edge = (const float*)d_in[13];
  const float* b_ln_edge = (const float*)d_in[14];
  const float* wq        = (const float*)d_in[15];
  const float* bq        = (const float*)d_in[16];
  const float* wk        = (const float*)d_in[17];
  const float* bk        = (const float*)d_in[18];
  const float* wv        = (const float*)d_in[19];
  const float* bv        = (const float*)d_in[20];
  const float* wo        = (const float*)d_in[21];
  const float* bo        = (const float*)d_in[22];
  const float* g_ln2     = (const float*)d_in[23];
  const float* b_ln2     = (const float*)d_in[24];
  const float* w_ff1     = (const float*)d_in[25];
  const float* b_ff1     = (const float*)d_in[26];
  const float* w_ff2     = (const float*)d_in[27];
  const float* b_ff2     = (const float*)d_in[28];
  const float* g_ln3     = (const float*)d_in[29];
  const float* b_ln3     = (const float*)d_in[30];

  char* ws = (char*)d_ws;
  _Float16* mem16  = (_Float16*)(ws + OFF_MEM16);
  _Float16* wE16   = (_Float16*)(ws + OFF_WE);
  _Float16* wP16   = (_Float16*)(ws + OFF_WP);
  float*    node_s = (float*)(ws + OFF_NS);
  float*    node_t = (float*)(ws + OFF_NT);
  float*    qkbuf  = (float*)(ws + OFF_QK);
  float*    sbbuf  = (float*)(ws + OFF_SB);
  float*    scbuf  = (float*)(ws + OFF_SC);
  float*    ambuf  = (float*)(ws + OFF_AM);

  float* xout     = (float*)d_out;
  float* edge_out = (float*)d_out + (size_t)NN*DM;

  swz_kernel<<<32, 256, 0, stream>>>(w_mem_e, wE16);
  swz_kernel<<<32, 256, 0, stream>>>(w_pe,    wP16);
  prep_kernel<<<NN, 128, 0, stream>>>(node, w_mem_s, w_mem_t, wq, bq, wk, bk,
                                      node_s, node_t, qkbuf, sbbuf);
  fuse_edge_kernel<<<(NN*NN)/64, 128, 0, stream>>>(
      edge, wE16, wP16, node_s, node_t,
      b_mem, g_ln_mem, b_ln_mem, b_pe, g_ln_pe, b_ln_pe, g_ln_edge, b_ln_edge,
      mem16, edge_out);
  scores_kernel<<<NN, 256, 0, stream>>>(mem16, qkbuf, sbbuf, emask, scbuf);
  softmax_kernel<<<NN*HH, 256, 0, stream>>>(scbuf);
  am_kernel<<<NN, 128, 0, stream>>>(mem16, scbuf, ambuf);
  final_kernel<<<NN, 256, 0, stream>>>(ambuf, wv, bv, wo, bo, node, g_ln2, b_ln2,
                                       w_ff1, b_ff1, w_ff2, b_ff2, g_ln3, b_ln3, xout);
}
